// FeatureEncodingLayer_438086664765
// MI455X (gfx1250) — matrix operand order repaired and checked by reading, not compiled, not run on hardware
//
#include <hip/hip_runtime.h>

#define EPSF   1e-5f
#define SIGMA  0.1f
#define BB     8
#define NN     4096
#define CIN    64
#define KK     32
#define C2     128
#define WOUT   16

typedef __attribute__((ext_vector_type(16))) __bf16 v16bf;
typedef __attribute__((ext_vector_type(8)))  float  v8f;
typedef __attribute__((ext_vector_type(4)))  int    v4i;

union FragBF { v16bf v; unsigned int u[8]; unsigned short s[16]; };

__device__ __forceinline__ unsigned short f2bf(float f) {
  unsigned int x = __float_as_uint(f);
  x += 0x7FFFu + ((x >> 16) & 1u);          // round-to-nearest-even
  return (unsigned short)(x >> 16);
}
__device__ __forceinline__ float bf2f(unsigned short s) {
  return __uint_as_float(((unsigned int)s) << 16);
}
// packed f32x2 -> bf16x2 (v_cvt_pk_bf16_f32 when available)
__device__ __forceinline__ unsigned int pack2bf(float a, float b) {
#if __has_builtin(__builtin_amdgcn_cvt_pk_bf16_f32)
  typedef __attribute__((ext_vector_type(2))) __bf16 v2bf;
  union { v2bf v; unsigned int u; } cv;
  cv.v = __builtin_amdgcn_cvt_pk_bf16_f32(a, b);
  return cv.u;
#else
  return (unsigned int)f2bf(a) | ((unsigned int)f2bf(b) << 16);
#endif
}

// ---- CDNA5 async global->LDS copy (ASYNCcnt path), with safe fallback ----
#if __has_builtin(__builtin_amdgcn_global_load_async_to_lds_b128)
#define HAVE_ASYNC_LDS 1
typedef __attribute__((address_space(1))) v4i gas_v4i;
typedef __attribute__((address_space(3))) v4i las_v4i;
__device__ __forceinline__ void async_cp16(void* g, void* l) {
  __builtin_amdgcn_global_load_async_to_lds_b128(
      (gas_v4i*)g, (las_v4i*)l, 0, 0);
}
__device__ __forceinline__ void async_wait0() {
#if __has_builtin(__builtin_amdgcn_s_wait_asynccnt)
  __builtin_amdgcn_s_wait_asynccnt(0);
#else
  asm volatile("s_wait_asynccnt 0" ::: "memory");
#endif
}
#endif

// ---------------- workspace layout (bytes) ----------------
#define IDX_OFF 0u                 // int   [8*4096*32]            4 MB
#define DS_OFF  4194304u           // float [32768]
#define GF_OFF  4325376u           // bf16  [32768*128]            8 MB
#define W1_OFF  12713984u          // bf16  [64*64]
#define W2_OFF  12722176u          // bf16  [128*64]
#define LW_OFF  12738560u          // bf16  [128*2048]             512 KB
#define PB_OFF  13262848u          // float [665] folded params
// param block float offsets:
// t1:0(64) t2:64(128) tf:192(128)
// w0f:320(24) b0f:344(8) w1f:352(64) b1f:416(8) w2f:424(128) b2f:552(16)
// d0W:568(8) d0b:576(8) d1W:584(64) d1b:648(8) d2W:656(8) d2b:664(1)

struct PtrTab { const float* p[57]; };

// =====================================================================
// Kernel 1: fold BN into weights, convert GEMM weights to bf16
// =====================================================================
__global__ __launch_bounds__(256) void prep_kernel(PtrTab T, unsigned short* w1bf,
    unsigned short* w2bf, unsigned short* lwbf, float* pb)
{
  int tid = threadIdx.x;
  { // feature MLP layer 1 (64->64), inputs 38..43 = (W,b,g,bt,m,v)
    const float *W=T.p[38], *b=T.p[39], *g=T.p[40], *bt=T.p[41], *m=T.p[42], *v=T.p[43];
    for (int i = tid; i < 64*64; i += 256) {
      int n = i >> 6;
      w1bf[i] = f2bf(g[n]*rsqrtf(v[n]+EPSF) * W[i]);
    }
    if (tid < 64) {
      float s = g[tid]*rsqrtf(v[tid]+EPSF);
      pb[tid] = s*(b[tid]-m[tid]) + bt[tid];
    }
  }
  { // feature MLP layer 2 (64->128), inputs 44..49
    const float *W=T.p[44], *b=T.p[45], *g=T.p[46], *bt=T.p[47], *m=T.p[48], *v=T.p[49];
    for (int i = tid; i < 128*64; i += 256) {
      int n = i >> 6;
      w2bf[i] = f2bf(g[n]*rsqrtf(v[n]+EPSF) * W[i]);
    }
    if (tid < 128) {
      float s = g[tid]*rsqrtf(v[tid]+EPSF);
      pb[64+tid] = s*(b[tid]-m[tid]) + bt[tid];
    }
  }
  { // final linear (2048->128) + bnlin; 50=lin_W 51=lin_b 52..55=(g,bt,m,v)
    const float *W=T.p[50], *lb=T.p[51], *g=T.p[52], *bt=T.p[53], *m=T.p[54], *v=T.p[55];
    for (int i = tid; i < 128*2048; i += 256) {
      int o = i >> 11;
      lwbf[i] = f2bf(g[o]*rsqrtf(v[o]+EPSF) * W[i]);
    }
    if (tid < 128) {
      float s = g[tid]*rsqrtf(v[tid]+EPSF);
      pb[192+tid] = s*(lb[tid]-m[tid]) + bt[tid];
    }
  }
  if (tid < 8) { // wnet layer0 (3->8), inputs 20..25
    const float *W=T.p[20], *b=T.p[21], *g=T.p[22], *bt=T.p[23], *m=T.p[24], *v=T.p[25];
    float s = g[tid]*rsqrtf(v[tid]+EPSF);
    for (int c = 0; c < 3; ++c) pb[320 + tid*3 + c] = s * W[tid*3 + c];
    pb[344 + tid] = s*(b[tid]-m[tid]) + bt[tid];
  }
  if (tid < 8) { // wnet layer1 (8->8), inputs 26..31
    const float *W=T.p[26], *b=T.p[27], *g=T.p[28], *bt=T.p[29], *m=T.p[30], *v=T.p[31];
    float s = g[tid]*rsqrtf(v[tid]+EPSF);
    for (int c = 0; c < 8; ++c) pb[352 + tid*8 + c] = s * W[tid*8 + c];
    pb[416 + tid] = s*(b[tid]-m[tid]) + bt[tid];
  }
  if (tid < 16) { // wnet layer2 (8->16), inputs 32..37
    const float *W=T.p[32], *b=T.p[33], *g=T.p[34], *bt=T.p[35], *m=T.p[36], *v=T.p[37];
    float s = g[tid]*rsqrtf(v[tid]+EPSF);
    for (int c = 0; c < 8; ++c) pb[424 + tid*8 + c] = s * W[tid*8 + c];
    pb[552 + tid] = s*(b[tid]-m[tid]) + bt[tid];
  }
  if (tid < 8) { // density layer0 (1->8), inputs 2..7
    const float *W=T.p[2], *b=T.p[3], *g=T.p[4], *bt=T.p[5], *m=T.p[6], *v=T.p[7];
    float s = g[tid]*rsqrtf(v[tid]+EPSF);
    pb[568 + tid] = s * W[tid];
    pb[576 + tid] = s*(b[tid]-m[tid]) + bt[tid];
  }
  if (tid < 8) { // density layer1 (8->8), inputs 8..13
    const float *W=T.p[8], *b=T.p[9], *g=T.p[10], *bt=T.p[11], *m=T.p[12], *v=T.p[13];
    float s = g[tid]*rsqrtf(v[tid]+EPSF);
    for (int c = 0; c < 8; ++c) pb[584 + tid*8 + c] = s * W[tid*8 + c];
    pb[648 + tid] = s*(b[tid]-m[tid]) + bt[tid];
  }
  if (tid == 0) { // density layer2 (8->1), inputs 14..19
    const float *W=T.p[14], *b=T.p[15], *g=T.p[16], *bt=T.p[17], *m=T.p[18], *v=T.p[19];
    float s = g[0]*rsqrtf(v[0]+EPSF);
    for (int c = 0; c < 8; ++c) pb[656 + c] = s * W[c];
    pb[664] = s*(b[0]-m[0]) + bt[0];
  }
}

// =====================================================================
// Kernel 2: brute-force kNN (k=32) + kernel density + density MLP
// =====================================================================
__global__ __launch_bounds__(128) void knn_dens_kernel(
    const float* __restrict__ xyz, int* __restrict__ idx,
    float* __restrict__ ds, const float* __restrict__ pb)
{
  __shared__ float sx[128], sy[128], sz[128];
  int b = blockIdx.y;
  int q = blockIdx.x * 128 + threadIdx.x;
  const float* X = xyz + (size_t)b * 3 * NN;
  float qx = X[q], qy = X[NN + q], qz = X[2*NN + q];

  float bd[KK]; int bi[KK];
#pragma unroll
  for (int j = 0; j < KK; ++j) { bd[j] = 3.4e38f; bi[j] = 0; }
  float dens = 0.f;
  const float inv2s2 = 1.0f / (2.0f * SIGMA * SIGMA);

  for (int tile = 0; tile < NN; tile += 128) {
    __syncthreads();
    sx[threadIdx.x] = X[tile + threadIdx.x];
    sy[threadIdx.x] = X[NN + tile + threadIdx.x];
    sz[threadIdx.x] = X[2*NN + tile + threadIdx.x];
    __syncthreads();
    for (int c = 0; c < 128; ++c) {
      float dx = qx - sx[c], dy = qy - sy[c], dz = qz - sz[c];
      float d = dx*dx + dy*dy + dz*dz;
      dens += __expf(-d * inv2s2);
      if (d < bd[KK-1]) {                 // rare: register insertion sort
        float cd = d; int ci = tile + c;
#pragma unroll
        for (int j = 0; j < KK; ++j) {
          if (cd < bd[j]) {
            float td = bd[j]; int ti = bi[j];
            bd[j] = cd; bi[j] = ci; cd = td; ci = ti;
          }
        }
      }
    }
  }
  dens = dens * (1.0f / (float)NN) * (1.0f / (2.5f * SIGMA));

  const float* d0W = pb + 568; const float* d0b = pb + 576;
  const float* d1W = pb + 584; const float* d1b = pb + 648;
  const float* d2W = pb + 656; const float* d2b = pb + 664;
  float h0[8], h1[8];
#pragma unroll
  for (int j = 0; j < 8; ++j) h0[j] = fmaxf(d0W[j]*dens + d0b[j], 0.f);
#pragma unroll
  for (int j = 0; j < 8; ++j) {
    float a = d1b[j];
#pragma unroll
    for (int i = 0; i < 8; ++i) a += d1W[j*8+i]*h0[i];
    h1[j] = fmaxf(a, 0.f);
  }
  float a2 = d2b[0];
#pragma unroll
  for (int i = 0; i < 8; ++i) a2 += d2W[i]*h1[i];

  size_t gq = (size_t)b * NN + q;
  ds[gq] = fmaxf(a2, 0.f);
#pragma unroll
  for (int j = 0; j < KK; ++j) idx[gq*KK + j] = bi[j];
}

// =====================================================================
// Kernel 3: pointwise feature MLP (64->64->128) via bf16 WMMA
// =====================================================================
__global__ __launch_bounds__(256) void feat_mlp_kernel(
    const float* __restrict__ feat, const unsigned short* __restrict__ w1bf,
    const unsigned short* __restrict__ w2bf, const float* __restrict__ pb,
    unsigned short* __restrict__ gf)
{
  __shared__ unsigned short X1[8][16*64];  // per-wave bf16 hidden, 16 KB
  const float* t1 = pb + 0;
  const float* t2 = pb + 64;
  int wave = threadIdx.x >> 5, lane = threadIdx.x & 31;
  int g = lane >> 4, ml = lane & 15;
  int p0 = (blockIdx.x * 8 + wave) * 16;
  int b = p0 >> 12, n0 = p0 & (NN - 1);
  const float* F = feat + (size_t)b * CIN * NN;

  v8f zero = {0.f,0.f,0.f,0.f,0.f,0.f,0.f,0.f};

  // ---- layer 1 ----
  v8f acc[4];
#pragma unroll
  for (int t = 0; t < 4; ++t) acc[t] = zero;
  const unsigned int* W1u = (const unsigned int*)w1bf;
  for (int ks = 0; ks < 2; ++ks) {
    int kb = ks * 32;
    FragBF A;
#pragma unroll
    for (int v = 0; v < 8; ++v) {
      int k0 = kb + g*16 + v*2;
      A.u[v] = pack2bf(F[(size_t)k0*NN + n0 + ml],
                       F[(size_t)(k0+1)*NN + n0 + ml]);
    }
#pragma unroll
    for (int t = 0; t < 4; ++t) {
      FragBF Bf;
#pragma unroll
      for (int v = 0; v < 8; ++v)
        Bf.u[v] = W1u[(t*16 + ml)*32 + (kb>>1) + g*8 + v];
      acc[t] = __builtin_amdgcn_wmma_f32_16x16x32_bf16(
          false, A.v, false, Bf.v, (short)0, acc[t], false, false);
    }
  }
#pragma unroll
  for (int t = 0; t < 4; ++t) {
    int ch = t*16 + ml;
#pragma unroll
    for (int r = 0; r < 8; ++r) {
      int m = r + 8*g;
      X1[wave][m*64 + ch] = f2bf(fmaxf(acc[t][r] + t1[ch], 0.f));
    }
  }
  __syncthreads();

  // ---- layer 2 ----
  v8f acc2[8];
#pragma unroll
  for (int t = 0; t < 8; ++t) acc2[t] = zero;
  const unsigned int* X1u = (const unsigned int*)&X1[wave][0];
  const unsigned int* W2u = (const unsigned int*)w2bf;
  for (int ks = 0; ks < 2; ++ks) {
    int kb = ks * 32;
    FragBF A;
#pragma unroll
    for (int v = 0; v < 8; ++v) {
      int k0 = kb + g*16 + v*2;
      A.u[v] = X1u[(ml*64 + k0) >> 1];
    }
#pragma unroll
    for (int t = 0; t < 8; ++t) {
      FragBF Bf;
#pragma unroll
      for (int v = 0; v < 8; ++v)
        Bf.u[v] = W2u[(t*16 + ml)*32 + (kb>>1) + g*8 + v];
      acc2[t] = __builtin_amdgcn_wmma_f32_16x16x32_bf16(
          false, A.v, false, Bf.v, (short)0, acc2[t], false, false);
    }
  }
#pragma unroll
  for (int t = 0; t < 8; ++t) {
    int ch = t*16 + ml;
#pragma unroll
    for (int r = 0; r < 8; ++r) {
      int m = r + 8*g;
      gf[(size_t)(p0 + m)*C2 + ch] = f2bf(fmaxf(acc2[t][r] + t2[ch], 0.f));
    }
  }
}

// =====================================================================
// Kernel 4 (fused): async gather -> LDS + density-scaled wnet + per-point
// aggregation GEMM + final 2048->128 linear, all bf16 WMMA.
// block = 128 threads (4 waves) handles 16 points.
// LDS: sA[16][32][128] bf16 (raw gathered rows, 128K) | sW[16][16][32] bf16
//      (density-scaled wnet, 16K) | sAgg[16][2048] bf16 (64K)
// =====================================================================
__global__ __launch_bounds__(128) void agg_final_kernel(
    const float* __restrict__ xyz, const int* __restrict__ idx,
    const float* __restrict__ ds, const unsigned short* __restrict__ gf,
    const unsigned short* __restrict__ lwbf, const float* __restrict__ pb,
    float* __restrict__ out)
{
  extern __shared__ unsigned char smem[];
  unsigned short* sA   = (unsigned short*)smem;              // [16][32][128] bf16
  unsigned short* sW   = (unsigned short*)(smem + 131072);   // [16][16][32]  bf16
  unsigned short* sAgg = (unsigned short*)(smem + 147456);   // [16][2048]    bf16

  const float* tf  = pb + 192;
  const float* w0f = pb + 320; const float* b0f = pb + 344;
  const float* w1f = pb + 352; const float* b1f = pb + 416;
  const float* w2f = pb + 424; const float* b2f = pb + 552;

  int tid = threadIdx.x;
  int ptbase = blockIdx.x * 16;
  int b = ptbase >> 12, n0 = ptbase & (NN - 1);
  const float* X = xyz + (size_t)b * 3 * NN;

  // ---- stage 1: async copy raw bf16 rows to LDS; wnet scaled by density ----
  for (int i = 0; i < 4; ++i) {
    int p  = tid + i*128;
    int pt = p >> 5, k = p & 31;
    int n  = n0 + pt;
    int nb = idx[((size_t)b*NN + n)*KK + k];
    unsigned char* grow = (unsigned char*)(gf + (size_t)(b*NN + nb)*C2);
    unsigned char* lrow = (unsigned char*)sA + ((size_t)(pt*KK + k))*C2*2;
#ifdef HAVE_ASYNC_LDS
#pragma unroll
    for (int j = 0; j < 16; ++j) async_cp16(grow + j*16, lrow + j*16);
#else
#pragma unroll
    for (int j = 0; j < 16; ++j) ((uint4*)lrow)[j] = ((const uint4*)grow)[j];
#endif
    // wnet(rel_xyz): 3 -> 8 -> 8 -> 16 (BN folded); fold density into w
    float sc = ds[(size_t)b*NN + nb];
    float rx = X[nb] - X[n];
    float ry = X[NN + nb] - X[NN + n];
    float rz = X[2*NN + nb] - X[2*NN + n];
    float h0[8], h1[8];
#pragma unroll
    for (int j = 0; j < 8; ++j)
      h0[j] = fmaxf(w0f[j*3]*rx + w0f[j*3+1]*ry + w0f[j*3+2]*rz + b0f[j], 0.f);
#pragma unroll
    for (int j = 0; j < 8; ++j) {
      float a = b1f[j];
#pragma unroll
      for (int q = 0; q < 8; ++q) a += w1f[j*8+q]*h0[q];
      h1[j] = fmaxf(a, 0.f);
    }
#pragma unroll
    for (int j = 0; j < 16; ++j) {
      float a = b2f[j];
#pragma unroll
      for (int q = 0; q < 8; ++q) a += w2f[j*8+q]*h1[q];
      sW[((size_t)pt*16 + j)*KK + k] = f2bf(sc * fmaxf(a, 0.f));
    }
  }
#ifdef HAVE_ASYNC_LDS
  async_wait0();
#endif
  __syncthreads();

  int wave = tid >> 5, lane = tid & 31;
  int g = lane >> 4, ml = lane & 15;
  v8f zero = {0.f,0.f,0.f,0.f,0.f,0.f,0.f,0.f};

  // ---- stage 2: aggT[w,c] = sum_k (ds*w)[k,w] * gf[k,c]  (A=w^T, B=gf) ----
  const unsigned int* sWu = (const unsigned int*)sW;
  for (int pp = 0; pp < 4; ++pp) {
    int pt = wave*4 + pp;
    FragBF A;
#pragma unroll
    for (int v = 0; v < 8; ++v) {
      int k0 = g*16 + v*2;
      A.u[v] = sWu[((pt*16 + ml)*KK + k0) >> 1];
    }
    for (int t = 0; t < 8; ++t) {
      FragBF Bf;
#pragma unroll
      for (int v = 0; v < 8; ++v) {
        int k0 = g*16 + 2*v;
        int c  = t*16 + ml;
        Bf.s[2*v]   = sA[(size_t)(pt*KK + k0    )*C2 + c];
        Bf.s[2*v+1] = sA[(size_t)(pt*KK + k0 + 1)*C2 + c];
      }
      v8f acc = zero;
      acc = __builtin_amdgcn_wmma_f32_16x16x32_bf16(
          false, A.v, false, Bf.v, (short)0, acc, false, false);
#pragma unroll
      for (int r = 0; r < 8; ++r) {
        int w_ = r + 8*g;          // D row m = w index
        int c  = t*16 + ml;        // D col n = channel
        sAgg[(size_t)pt*2048 + c*16 + w_] = f2bf(acc[r]);
      }
    }
  }
  __syncthreads();

  // ---- stage 3: out = relu(agg(16x2048) @ linW'^T + tf), 2 n-tiles/wave ----
  const unsigned int* sGu = (const unsigned int*)sAgg;
  const unsigned int* Lu  = (const unsigned int*)lwbf;
  v8f acc0 = zero, acc1 = zero;
  for (int ks = 0; ks < 64; ++ks) {
    int kb = ks * 32;
    FragBF A;
#pragma unroll
    for (int v = 0; v < 8; ++v) {
      int k0 = kb + g*16 + v*2;
      A.u[v] = sGu[(ml*2048 + k0) >> 1];
    }
    {
      int o = (wave*2 + 0)*16 + ml;
      FragBF Bf;
#pragma unroll
      for (int v = 0; v < 8; ++v) Bf.u[v] = Lu[o*1024 + (kb>>1) + g*8 + v];
      acc0 = __builtin_amdgcn_wmma_f32_16x16x32_bf16(
          false, A.v, false, Bf.v, (short)0, acc0, false, false);
    }
    {
      int o = (wave*2 + 1)*16 + ml;
      FragBF Bf;
#pragma unroll
      for (int v = 0; v < 8; ++v) Bf.u[v] = Lu[o*1024 + (kb>>1) + g*8 + v];
      acc1 = __builtin_amdgcn_wmma_f32_16x16x32_bf16(
          false, A.v, false, Bf.v, (short)0, acc1, false, false);
    }
  }
#pragma unroll
  for (int j = 0; j < 2; ++j) {
    int o = (wave*2 + j)*16 + ml;
    v8f a = j ? acc1 : acc0;
#pragma unroll
    for (int r = 0; r < 8; ++r) {
      int m = r + 8*g;
      out[((size_t)(b*C2 + o))*NN + n0 + m] = fmaxf(a[r] + tf[o], 0.f);
    }
  }
}

// =====================================================================
extern "C" void kernel_launch(void* const* d_in, const int* in_sizes, int n_in,
                              void* d_out, int out_size, void* d_ws, size_t ws_size,
                              hipStream_t stream)
{
  (void)in_sizes; (void)out_size; (void)ws_size;
  const float* xyz  = (const float*)d_in[0];
  const float* feat = (const float*)d_in[1];

  unsigned char* ws = (unsigned char*)d_ws;
  int*            idx  = (int*)(ws + IDX_OFF);
  float*          ds   = (float*)(ws + DS_OFF);
  unsigned short* gf   = (unsigned short*)(ws + GF_OFF);
  unsigned short* w1bf = (unsigned short*)(ws + W1_OFF);
  unsigned short* w2bf = (unsigned short*)(ws + W2_OFF);
  unsigned short* lwbf = (unsigned short*)(ws + LW_OFF);
  float*          pb   = (float*)(ws + PB_OFF);
  float*          out  = (float*)d_out;

  // output[0:98304] = xyz passthrough
  (void)hipMemcpyAsync(d_out, d_in[0], (size_t)BB*3*NN*sizeof(float),
                       hipMemcpyDeviceToDevice, stream);

  PtrTab tab;
  for (int i = 0; i < 57 && i < n_in; ++i) tab.p[i] = (const float*)d_in[i];

  hipLaunchKernelGGL(prep_kernel, dim3(1), dim3(256), 0, stream,
                     tab, w1bf, w2bf, lwbf, pb);

  hipLaunchKernelGGL(knn_dens_kernel, dim3(NN/128, BB), dim3(128), 0, stream,
                     xyz, idx, ds, pb);

  hipLaunchKernelGGL(feat_mlp_kernel, dim3((BB*NN)/128), dim3(256), 0, stream,
                     feat, w1bf, w2bf, pb, gf);

  // idempotent, deterministic attribute raise for 208 KB dynamic LDS
  (void)hipFuncSetAttribute(reinterpret_cast<const void*>(agg_final_kernel),
                            hipFuncAttributeMaxDynamicSharedMemorySize, 212992);
  hipLaunchKernelGGL(agg_final_kernel, dim3((BB*NN)/16), dim3(128), 212992, stream,
                     xyz, idx, ds, gf, lwbf, pb, out + (size_t)BB*3*NN);
}
